// LinearList_7172595384655
// MI455X (gfx1250) — matrix operand order repaired and checked by reading, not compiled, not run on hardware
//
#include <hip/hip_runtime.h>

#define PER_IN      128
#define PER_OUT     128
#define NUM_LINEARS 64
#define BATCH       8192
#define ROW_STRIDE  (PER_IN * NUM_LINEARS)   // 8192 floats per row of input/weight/output
#define BM          128                      // batch rows per block
#define LDX         136                      // padded LDS row stride (bf16 elems), 272B: bank-rotating, 16B aligned
#define LDW         136

typedef __attribute__((ext_vector_type(16))) __bf16 v16bf;
typedef __attribute__((ext_vector_type(8)))  float  v8f;

union FragAB {
    uint4  q[2];
    v16bf  v;
};

// f32 -> bf16, round-to-nearest-even
__device__ __forceinline__ unsigned short f2bf(float f) {
    union { float f; unsigned int u; } x;
    x.f = f;
    unsigned int u = x.u;
    u += 0x7FFFu + ((u >> 16) & 1u);
    return (unsigned short)(u >> 16);
}

__global__ __launch_bounds__(256) void grouped_linear_wmma_bf16(
    const float* __restrict__ in,   // [BATCH, 8192]
    const float* __restrict__ w,    // [PER_OUT, 8192]
    float* __restrict__ out)        // [BATCH, 8192]
{
    __shared__ __attribute__((aligned(16))) unsigned short Xs[BM * LDX];      // X tile, bf16 bits
    __shared__ __attribute__((aligned(16))) unsigned short Ws[PER_OUT * LDW]; // W tile, bf16 bits

    const int tid  = threadIdx.x;
    const int mb   = blockIdx.x;          // batch tile
    const int g    = blockIdx.y;          // group
    const int row0 = mb * BM;
    const int col0 = g * PER_IN;

    // ---- stage X tile: 128x128 f32 -> bf16 in LDS (coalesced float4 per row) ----
    for (int idx = tid; idx < BM * (PER_IN / 4); idx += 256) {
        const int r  = idx >> 5;          // 32 float4 per 128-wide row
        const int cq = idx & 31;
        float4 v = *(const float4*)(in + (size_t)(row0 + r) * ROW_STRIDE + col0 + cq * 4);
        uint2 p;
        p.x = (unsigned)f2bf(v.x) | ((unsigned)f2bf(v.y) << 16);
        p.y = (unsigned)f2bf(v.z) | ((unsigned)f2bf(v.w) << 16);
        *(uint2*)&Xs[r * LDX + cq * 4] = p;
    }
    // ---- stage W tile (L2-resident source, 4MB total) ----
    for (int idx = tid; idx < PER_OUT * (PER_IN / 4); idx += 256) {
        const int r  = idx >> 5;
        const int cq = idx & 31;
        float4 v = *(const float4*)(w + (size_t)r * ROW_STRIDE + col0 + cq * 4);
        uint2 p;
        p.x = (unsigned)f2bf(v.x) | ((unsigned)f2bf(v.y) << 16);
        p.y = (unsigned)f2bf(v.z) | ((unsigned)f2bf(v.w) << 16);
        *(uint2*)&Ws[r * LDW + cq * 4] = p;
    }
    __syncthreads();

    const int wave  = tid >> 5;           // 0..7 : 16-row M strip per wave
    const int lane  = tid & 31;
    const int r16   = lane & 15;
    const int hi    = lane >> 4;          // half-wave select
    const int mbase = wave * 16;

    v8f acc[8] = {};                      // 8 N-tiles of 16 -> full N=128 per wave

    // ---- K loop: 4 steps of 32, 8 WMMA tiles along N per step ----
#pragma unroll
    for (int kk = 0; kk < PER_IN; kk += 32) {
        // A fragment (16x32 bf16): lane r16 = M row; elems 0..7 -> K=kk+hi*8..,
        // elems 8..15 -> K=kk+16+hi*8.. (per ISA 16-bit A layout)
        FragAB a;
        const unsigned short* ab = &Xs[(mbase + r16) * LDX + kk + hi * 8];
        a.q[0] = *(const uint4*)(ab);
        a.q[1] = *(const uint4*)(ab + 16);

#pragma unroll
        for (int n = 0; n < 8; ++n) {
            // B fragment (32x16 bf16): lane r16 = N col; elems 0..7 -> K=kk+hi*8..,
            // elems 8..15 -> K=kk+16+hi*8.. (same K order as A)
            FragAB b;
            const unsigned short* bb = &Ws[(n * 16 + r16) * LDW + kk + hi * 8];
            b.q[0] = *(const uint4*)(bb);
            b.q[1] = *(const uint4*)(bb + 16);

            acc[n] = __builtin_amdgcn_wmma_f32_16x16x32_bf16(
                false, a.v, false, b.v, (short)0, acc[n], false, false);
        }
    }

    // ---- store D: VGPR rr holds M = rr (lanes 0-15) / 8+rr (lanes 16-31), N = r16 ----
    float* obase = out + (size_t)(row0 + mbase + hi * 8) * ROW_STRIDE + g * PER_OUT + r16;
#pragma unroll
    for (int n = 0; n < 8; ++n) {
#pragma unroll
        for (int rr = 0; rr < 8; ++rr) {
            obase[(size_t)rr * ROW_STRIDE + n * 16] = acc[n][rr];
        }
    }
}

extern "C" void kernel_launch(void* const* d_in, const int* in_sizes, int n_in,
                              void* d_out, int out_size, void* d_ws, size_t ws_size,
                              hipStream_t stream) {
    const float* in = (const float*)d_in[0];   // [8192, 8192] f32
    const float* w  = (const float*)d_in[1];   // [128, 8192]  f32
    float* out      = (float*)d_out;           // [8192, 8192] f32

    dim3 grid(BATCH / BM, NUM_LINEARS);        // (64, 64)
    grouped_linear_wmma_bf16<<<grid, 256, 0, stream>>>(in, w, out);
}
